// SplineLinear_70772471103879
// MI455X (gfx1250) — matrix operand order repaired and checked by reading, not compiled, not run on hardware
//
#include <hip/hip_runtime.h>
#include <math.h>

#define K_CP   8
#define OUT_F  4096
#define IN_F   4096
#define N_ROWS 8192

#define BM 128
#define BN 256
#define BK 32
#define LDT 40   // padded LDS row stride in bf16 elements (80 B, keeps 16B alignment)

typedef __attribute__((ext_vector_type(16))) __bf16 v16bf;
typedef __attribute__((ext_vector_type(8)))  __bf16 v8bf;
typedef __attribute__((ext_vector_type(4)))  __bf16 v4bf;
typedef __attribute__((ext_vector_type(8)))  float  v8f;
typedef __attribute__((ext_vector_type(4)))  float  v4f;
typedef __attribute__((ext_vector_type(4)))  unsigned int v4u;

static __device__ inline v8bf pack8(v4f x, v4f y) {
    v8bf r;
    r[0] = (__bf16)x[0]; r[1] = (__bf16)x[1]; r[2] = (__bf16)x[2]; r[3] = (__bf16)x[3];
    r[4] = (__bf16)y[0]; r[5] = (__bf16)y[1]; r[6] = (__bf16)y[2]; r[7] = (__bf16)y[3];
    return r;
}

// ---------------------------------------------------------------------------
// Kernel 1: W_bf16[o,i] = CatmullRom(control_points)(o,i) + residual[o,i]
// One thread per 4 consecutive i (vectorized b128 loads, b64 stores).
// ---------------------------------------------------------------------------
__global__ __launch_bounds__(256) void build_weight_kernel(
    const float* __restrict__ cp,       // [K_CP+1, IN_F]
    const float* __restrict__ residual, // [OUT_F, IN_F]
    __bf16* __restrict__ wout)          // [OUT_F, IN_F]
{
    const int idx  = blockIdx.x * blockDim.x + threadIdx.x;
    const int base = idx * 4;
    const int o = base >> 12;      // / IN_F
    const int i = base & (IN_F - 1);

    float ts = (float)o * ((float)K_CP / (float)(OUT_F - 1));
    int j = (int)floorf(ts);
    j = j < 1 ? 1 : (j > K_CP - 2 ? K_CP - 2 : j);
    float t  = ts - (float)j;      // clip before subtract, matching reference
    float t2 = t * t;
    float t3 = t2 * t;
    float c0 = -0.5f * t3 +        t2 - 0.5f * t;
    float c1 =  1.5f * t3 - 2.5f * t2 + 1.0f;
    float c2 = -1.5f * t3 + 2.0f * t2 + 0.5f * t;
    float c3 =  0.5f * t3 - 0.5f * t2;

    const float* cpb = cp + i;
    v4f m1 = *(const v4f*)(cpb + (j - 1) * IN_F);
    v4f p0 = *(const v4f*)(cpb + (j    ) * IN_F);
    v4f p1 = *(const v4f*)(cpb + (j + 1) * IN_F);
    v4f p2 = *(const v4f*)(cpb + (j + 2) * IN_F);
    v4f rs = *(const v4f*)(residual + base);

    v4f w = c0 * m1 + c1 * p0 + c2 * p1 + c3 * p2 + rs;

    v4bf wb;
    wb[0] = (__bf16)w[0]; wb[1] = (__bf16)w[1];
    wb[2] = (__bf16)w[2]; wb[3] = (__bf16)w[3];
    *(v4bf*)(wout + base) = wb;
}

// ---------------------------------------------------------------------------
// Kernel 2: out[n,o] = sum_i A[n,i] * W[o,i] + bias[o]
// 256 threads = 8 wave32. Block tile 128(M) x 256(N), K-step 32.
// Wave grid 2(M) x 4(N); each wave: 4x4 subtiles of 16x16 -> 16 WMMA / K-step
// against 16 ds_load_b128 (1:1 density). Double-buffered LDS, one barrier
// per K-step, global->register prefetch overlapped with WMMA.
// ---------------------------------------------------------------------------
__global__ __launch_bounds__(256) void spline_gemm_kernel(
    const float*  __restrict__ A,    // [N_ROWS, IN_F] fp32
    const __bf16* __restrict__ W,    // [OUT_F, IN_F]  bf16
    const float*  __restrict__ bias, // [OUT_F]
    float*        __restrict__ out)  // [N_ROWS, OUT_F]
{
    __shared__ __bf16 lA[2][BM * LDT];   // 2 x 10240 B
    __shared__ __bf16 lW[2][BN * LDT];   // 2 x 20480 B

    const int tid  = threadIdx.x;
    const int lane = tid & 31;
    const int half = lane >> 4;   // 0..1
    const int lr   = lane & 15;   // 0..15
    const int wave = tid >> 5;    // 0..7
    const int wm   = wave & 1;    // 0..1  (64 rows each)
    const int wn   = wave >> 1;   // 0..3  (64 cols each)

    const int blockM = blockIdx.y * BM;
    const int blockN = blockIdx.x * BN;

    // A fill: 2 threads per row, 16 fp32 each
    const int ldRowA = tid >> 1;          // 0..127
    const int ldColA = (tid & 1) * 16;    // 0 or 16
    // W fill: 1 thread per row, 32 bf16 each
    const int ldRowW = tid;               // 0..255

    const float*  gA = A + (size_t)(blockM + ldRowA) * IN_F + ldColA;
    const __bf16* gW = W + (size_t)(blockN + ldRowW) * IN_F;

    v8f acc[4][4];
#pragma unroll
    for (int ms = 0; ms < 4; ++ms)
#pragma unroll
        for (int ns = 0; ns < 4; ++ns)
            acc[ms][ns] = (v8f){0.f, 0.f, 0.f, 0.f, 0.f, 0.f, 0.f, 0.f};

    // prefetch K-step 0 into registers
    v4f a0 = *(const v4f*)(gA + 0);
    v4f a1 = *(const v4f*)(gA + 4);
    v4f a2 = *(const v4f*)(gA + 8);
    v4f a3 = *(const v4f*)(gA + 12);
    v4u w0 = *(const v4u*)(gW + 0);
    v4u w1 = *(const v4u*)(gW + 8);
    v4u w2 = *(const v4u*)(gW + 16);
    v4u w3 = *(const v4u*)(gW + 24);

    // stage into LDS buffer 0
    {
        __bf16* dA = &lA[0][ldRowA * LDT + ldColA];
        *(v8bf*)(dA)     = pack8(a0, a1);
        *(v8bf*)(dA + 8) = pack8(a2, a3);
        v4u* dW = (v4u*)&lW[0][ldRowW * LDT];
        dW[0] = w0; dW[1] = w1; dW[2] = w2; dW[3] = w3;
    }

    const int KT = IN_F / BK;  // 128
    for (int kt = 0; kt < KT; ++kt) {
        __syncthreads();
        const int cur = kt & 1;

        // prefetch next K-step (global -> registers) while computing
        const bool more = (kt + 1) < KT;
        if (more) {
            const float*  pA = gA + (kt + 1) * BK;
            const __bf16* pW = gW + (kt + 1) * BK;
            a0 = *(const v4f*)(pA + 0);
            a1 = *(const v4f*)(pA + 4);
            a2 = *(const v4f*)(pA + 8);
            a3 = *(const v4f*)(pA + 12);
            w0 = *(const v4u*)(pW + 0);
            w1 = *(const v4u*)(pW + 8);
            w2 = *(const v4u*)(pW + 16);
            w3 = *(const v4u*)(pW + 24);
        }

        // load fragments from LDS per the gfx1250 WMMA VGPR layouts
        v16bf afr[4];
#pragma unroll
        for (int ms = 0; ms < 4; ++ms) {
            const __bf16* p = &lA[cur][(wm * 64 + ms * 16 + lr) * LDT + 8 * half];
            v8bf lo = *(const v8bf*)(p);        // K = 8h+0..7
            v8bf hi = *(const v8bf*)(p + 16);   // K = 16+8h+0..7
#pragma unroll
            for (int e = 0; e < 8; ++e) { afr[ms][e] = lo[e]; afr[ms][8 + e] = hi[e]; }
        }
        v16bf bfr[4];
#pragma unroll
        for (int ns = 0; ns < 4; ++ns) {
            const __bf16* p = &lW[cur][(wn * 64 + ns * 16 + lr) * LDT + 8 * half];
            v8bf lo = *(const v8bf*)(p);        // K = 8h+0..7
            v8bf hi = *(const v8bf*)(p + 16);   // K = 16+8h+0..7
#pragma unroll
            for (int e = 0; e < 8; ++e) { bfr[ns][e] = lo[e]; bfr[ns][8 + e] = hi[e]; }
        }

#pragma unroll
        for (int ms = 0; ms < 4; ++ms)
#pragma unroll
            for (int ns = 0; ns < 4; ++ns)
                acc[ms][ns] = __builtin_amdgcn_wmma_f32_16x16x32_bf16(
                    false, afr[ms], false, bfr[ns],
                    (short)0, acc[ms][ns], false, false);

        // stage prefetched data into the other LDS buffer
        if (more) {
            const int nxt = cur ^ 1;
            __bf16* dA = &lA[nxt][ldRowA * LDT + ldColA];
            *(v8bf*)(dA)     = pack8(a0, a1);
            *(v8bf*)(dA + 8) = pack8(a2, a3);
            v4u* dW = (v4u*)&lW[nxt][ldRowW * LDT];
            dW[0] = w0; dW[1] = w1; dW[2] = w2; dW[3] = w3;
        }
    }

    // Epilogue: D element (VGPR v, lane) = (row v+8h, col lr). Coalesced
    // 64 B per half-wave stores, bias fused.
#pragma unroll
    for (int ms = 0; ms < 4; ++ms) {
#pragma unroll
        for (int ns = 0; ns < 4; ++ns) {
            const int col = blockN + wn * 64 + ns * 16 + lr;
            const float b = bias[col];
#pragma unroll
            for (int v = 0; v < 8; ++v) {
                const int row = blockM + wm * 64 + ms * 16 + v + 8 * half;
                out[(size_t)row * OUT_F + col] = acc[ms][ns][v] + b;
            }
        }
    }
}

// ---------------------------------------------------------------------------
extern "C" void kernel_launch(void* const* d_in, const int* in_sizes, int n_in,
                              void* d_out, int out_size, void* d_ws, size_t ws_size,
                              hipStream_t stream) {
    const float* input    = (const float*)d_in[0]; // [N_ROWS, IN_F]
    const float* cp       = (const float*)d_in[1]; // [K_CP+1, IN_F]
    const float* residual = (const float*)d_in[2]; // [OUT_F, IN_F]
    const float* bias     = (const float*)d_in[3]; // [OUT_F]
    float* out = (float*)d_out;                    // [N_ROWS, OUT_F]

    __bf16* wbf = (__bf16*)d_ws;                   // 4096*4096*2 = 33.5 MB

    {
        const int nthreads = OUT_F * IN_F / 4;
        build_weight_kernel<<<nthreads / 256, 256, 0, stream>>>(cp, residual, wbf);
    }
    {
        dim3 grid(OUT_F / BN, N_ROWS / BM);
        spline_gemm_kernel<<<grid, dim3(256), 0, stream>>>(input, wbf, bias, out);
    }
}
